// GATModel_13211319402609
// MI455X (gfx1250) — compile-verified, matrix operand order checked by reading, not run on hardware
//
#include <hip/hip_runtime.h>
#include <hip/hip_bf16.h>

// ---------------- problem constants ----------------
#define NN      5000            // nodes
#define NE      40000           // edges (w/o self loops)
#define ET      45000           // edges + self loops
#define INCH    2000
#define NH1     16
#define NC1     256
#define D1      4096            // NH1*NC1
#define OUTC    30
#define KP1     2016            // INCH padded to 32
#define MP      5120            // nodes padded to 128
#define NP2     32              // OUTC padded
#define NEG_SLOPE 0.2f
#define NT1     (KP1 / 32)      // 63 K-steps for GEMM1

typedef _Float16 v8h  __attribute__((ext_vector_type(8)));
typedef _Float16 v16h __attribute__((ext_vector_type(16)));
typedef float    v8f  __attribute__((ext_vector_type(8)));
typedef unsigned int tdm_v4u __attribute__((ext_vector_type(4)));
typedef int          tdm_v4i __attribute__((ext_vector_type(4)));
typedef int          tdm_v8i __attribute__((ext_vector_type(8)));

#if defined(__has_builtin)
#if __has_builtin(__builtin_amdgcn_tensor_load_to_lds) && __has_builtin(__builtin_amdgcn_s_wait_tensorcnt)
#define USE_TDM 1
#endif
#endif

// ---------------- helpers ----------------
__device__ __forceinline__ unsigned fenc(float f) {
    unsigned u = __float_as_uint(f);
    return (u & 0x80000000u) ? ~u : (u | 0x80000000u);
}
__device__ __forceinline__ float fdec(unsigned u) {
    return (u & 0x80000000u) ? __uint_as_float(u & 0x7FFFFFFFu) : __uint_as_float(~u);
}
__device__ __forceinline__ float elu1(float v) { return v > 0.f ? v : (__expf(v) - 1.f); }

// Build a 16-half WMMA fragment from an LDS row (32 contiguous halves per row).
// ISA 16-bit A/B layout: lanes 0-15 take K {0..7} then {16..23}; lanes 16-31
// take K {8..15} then {24..31}.  Two ds_load_b128 per fragment.
__device__ __forceinline__ v16h frag_from_lds(const _Float16* rowBase, int laneH) {
    const v8h lo = *reinterpret_cast<const v8h*>(rowBase + laneH * 8);
    const v8h hi = *reinterpret_cast<const v8h*>(rowBase + 16 + laneH * 8);
    v16h f;
#pragma unroll
    for (int i = 0; i < 8; ++i) { f[i] = lo[i]; f[i + 8] = hi[i]; }
    return f;
}

#ifdef USE_TDM
// Post one 2-D TDM tile load: global (row-major, rowStrideUnits in 8-byte units)
// -> LDS at byte offset lds_off.  Tile is tileD0 x tileD1 in 8-byte units.
// D# layout per CDNA5 ISA ch.10.8 (group0: count/type/lds/global addr,
// group1: data_size, dims, tile dims, stride; groups 2/3 zero => 2-D tensor).
__device__ __forceinline__ void tdm_load_2d(unsigned lds_off, const void* gptr,
                                            unsigned tileD0, unsigned tileD1,
                                            unsigned rowStrideUnits) {
    unsigned long long ga = (unsigned long long)(uintptr_t)gptr;
    tdm_v4u g0;
    g0[0] = 1u;                                              // count=1, user D#
    g0[1] = lds_off;                                         // lds_addr
    g0[2] = (unsigned)(ga & 0xFFFFFFFFu);                    // global_addr[31:0]
    g0[3] = (unsigned)((ga >> 32) & 0x01FFFFFFu) | (2u << 30); // addr[56:32] | type=2
    tdm_v8i g1;
    g1[0] = (int)(3u << 16);                                 // data_size = 3 (8B)
    g1[1] = (int)((tileD0 & 0xFFFFu) << 16);                 // tensor_dim0[15:0]
    g1[2] = (int)((tileD0 >> 16) | ((tileD1 & 0xFFFFu) << 16)); // d0 hi | tensor_dim1 lo
    g1[3] = (int)((tileD1 >> 16) | (tileD0 << 16));          // d1 hi | tile_dim0
    g1[4] = (int)(tileD1 & 0xFFFFu);                         // tile_dim1 | tile_dim2=0
    g1[5] = (int)rowStrideUnits;                             // tensor_dim0_stride lo32
    g1[6] = 0;                                               // stride0 hi | stride1 lo
    g1[7] = 0;                                               // stride1 hi
    tdm_v4i z = {0, 0, 0, 0};
#if __clang_major__ >= 23
    tdm_v8i z8 = {0, 0, 0, 0, 0, 0, 0, 0};
    __builtin_amdgcn_tensor_load_to_lds(g0, g1, z, z, z8, 0);
#else
    __builtin_amdgcn_tensor_load_to_lds(g0, g1, z, z, 0);
#endif
}
#endif

// ---------------- fill ----------------
__global__ void fill_u32_kernel(unsigned* __restrict__ p, unsigned v, unsigned n) {
    unsigned i = blockIdx.x * 256u + threadIdx.x;
    if (i < n) p[i] = v;
}

// ---------------- conversions ----------------
__global__ void convert_x_kernel(const float* __restrict__ x, _Float16* __restrict__ xf) {
    unsigned idx = blockIdx.x * 256u + threadIdx.x;
    if (idx >= (unsigned)MP * KP1) return;
    unsigned row = idx / KP1, col = idx - row * KP1;
    float v = (row < NN && col < INCH) ? x[(size_t)row * INCH + col] : 0.f;
    xf[idx] = (_Float16)v;
}
// W1 [INCH][D1] -> W1T f16 [D1][KP1]  (coalesced reads over n)
__global__ void convert_w1_kernel(const float* __restrict__ w, _Float16* __restrict__ wt) {
    unsigned idx = blockIdx.x * 256u + threadIdx.x;
    if (idx >= (unsigned)KP1 * D1) return;
    unsigned k = idx / D1, n = idx - k * D1;
    float v = (k < INCH) ? w[(size_t)k * D1 + n] : 0.f;
    wt[(size_t)n * KP1 + k] = (_Float16)v;
}
// W2 [D1][OUTC] -> W2T f16 [NP2][D1]
__global__ void convert_w2_kernel(const float* __restrict__ w, _Float16* __restrict__ wt) {
    unsigned idx = blockIdx.x * 256u + threadIdx.x;
    if (idx >= (unsigned)NP2 * D1) return;
    unsigned n = idx / D1, k = idx - n * D1;
    float v = (n < OUTC) ? w[(size_t)k * OUTC + n] : 0.f;
    wt[idx] = (_Float16)v;
}

// ---------------- GEMM1: h1[MP][D1] = xf16[MP][KP1] @ W1T[D1][KP1]^T ----------------
// Double-buffered LDS (2 x (A 8KB + B 4KB) dynamic).  TDM path: wave 0 posts
// tensor_load_to_lds descriptors for tile k+1 while all waves compute tile k;
// s_wait_tensorcnt(2) keeps exactly the newest descriptor pair in flight.
__global__ __launch_bounds__(256) void gemm1_kernel(const _Float16* __restrict__ A,
                                                    const _Float16* __restrict__ BT,
                                                    float* __restrict__ C) {
    extern __shared__ __align__(16) _Float16 smem[];   // 2 * 6144 halves
    const int tid   = threadIdx.x;
    const int lane  = tid & 31;
    const int w     = tid >> 5;
    const int wm    = w >> 1;          // 0..3 -> 32-row stripe
    const int wn    = w & 1;           // 0..1 -> 32-col stripe
    const int laneM = lane & 15;
    const int laneH = lane >> 4;
    const size_t gm = (size_t)blockIdx.y * 128;
    const size_t gn = (size_t)blockIdx.x * 64;

    v8f acc[2][2] = {};

#ifdef USE_TDM
    const bool issuer = (tid < 32);    // wave 0 drives the TDM
    if (issuer) {                      // prologue: tile 0 -> buffer 0
        tdm_load_2d(0u,    A  + gm * KP1, 8u, 128u, 504u);
        tdm_load_2d(8192u, BT + gn * KP1, 8u, 64u,  504u);
    }
#else
    const int arow = tid >> 1, acol = (tid & 1) * 16;
    const int brow = tid >> 2, bcol = (tid & 3) * 8;
    {   // prologue: tile 0 -> buffer 0 via VGPR staging
        const _Float16* ga = A + (gm + arow) * KP1 + acol;
        *reinterpret_cast<v8h*>(&smem[arow * 32 + acol])     = *reinterpret_cast<const v8h*>(ga);
        *reinterpret_cast<v8h*>(&smem[arow * 32 + acol + 8]) = *reinterpret_cast<const v8h*>(ga + 8);
        const _Float16* gb = BT + (gn + brow) * KP1 + bcol;
        *reinterpret_cast<v8h*>(&smem[4096 + brow * 32 + bcol]) = *reinterpret_cast<const v8h*>(gb);
    }
#endif

    for (int it = 0; it < NT1; ++it) {
        const int cb = it & 1;
        const int pb = cb ^ 1;
        const int ktNext = (it + 1) * 32;
#ifdef USE_TDM
        if (issuer) {
            if (it + 1 < NT1) {
                tdm_load_2d((unsigned)(pb * 12288),        A  + gm * KP1 + ktNext, 8u, 128u, 504u);
                tdm_load_2d((unsigned)(pb * 12288 + 8192), BT + gn * KP1 + ktNext, 8u, 64u,  504u);
                __builtin_amdgcn_s_wait_tensorcnt(2);   // current pair landed
            } else {
                __builtin_amdgcn_s_wait_tensorcnt(0);
            }
        }
#else
        if (it + 1 < NT1) {
            const _Float16* ga = A + (gm + arow) * KP1 + ktNext + acol;
            *reinterpret_cast<v8h*>(&smem[pb * 6144 + arow * 32 + acol])     = *reinterpret_cast<const v8h*>(ga);
            *reinterpret_cast<v8h*>(&smem[pb * 6144 + arow * 32 + acol + 8]) = *reinterpret_cast<const v8h*>(ga + 8);
            const _Float16* gb = BT + (gn + brow) * KP1 + ktNext + bcol;
            *reinterpret_cast<v8h*>(&smem[pb * 6144 + 4096 + brow * 32 + bcol]) = *reinterpret_cast<const v8h*>(gb);
        }
#endif
        __syncthreads();               // current buffer cb is ready for everyone
        const _Float16* As = smem + cb * 6144;
        const _Float16* Bs = As + 4096;
        v16h bf[2];
#pragma unroll
        for (int j = 0; j < 2; ++j)
            bf[j] = frag_from_lds(Bs + (wn * 32 + j * 16 + laneM) * 32, laneH);
#pragma unroll
        for (int i = 0; i < 2; ++i) {
            v16h af = frag_from_lds(As + (wm * 32 + i * 16 + laneM) * 32, laneH);
#pragma unroll
            for (int j = 0; j < 2; ++j)
                acc[i][j] = __builtin_amdgcn_wmma_f32_16x16x32_f16(
                    false, af, false, bf[j], (short)0, acc[i][j], false, false);
        }
        __syncthreads();               // reads of cb done -> safe to overwrite next round
    }
#pragma unroll
    for (int i = 0; i < 2; ++i)
#pragma unroll
        for (int j = 0; j < 2; ++j) {
            size_t row = gm + wm * 32 + i * 16 + 8 * laneH;
            size_t col = gn + wn * 32 + j * 16 + laneM;
#pragma unroll
            for (int r = 0; r < 8; ++r)
                C[(row + r) * D1 + col] = acc[i][j][r];
        }
}

// ---------------- GEMM2: h2[MP][NP2] = x1f16[MP][D1] @ W2T[NP2][D1]^T ----------------
__global__ __launch_bounds__(128) void gemm2_kernel(const _Float16* __restrict__ A,
                                                    const _Float16* __restrict__ BT,
                                                    float* __restrict__ C) {
    __shared__ __align__(16) _Float16 As[64][32];
    __shared__ __align__(16) _Float16 Bs[32][32];
    const int tid   = threadIdx.x;
    const int lane  = tid & 31;
    const int w     = tid >> 5;        // 0..3 -> 16-row stripe
    const int laneM = lane & 15;
    const int laneH = lane >> 4;
    const size_t gm = (size_t)blockIdx.y * 64;

    v8f acc[2] = {};
    const int arow = tid >> 1, acol = (tid & 1) * 16;
    const int brow = tid >> 2, bcol = (tid & 3) * 8;

    for (int kt = 0; kt < D1; kt += 32) {
        const _Float16* ga = A + (gm + arow) * D1 + kt + acol;
        *reinterpret_cast<v8h*>(&As[arow][acol])     = *reinterpret_cast<const v8h*>(ga);
        *reinterpret_cast<v8h*>(&As[arow][acol + 8]) = *reinterpret_cast<const v8h*>(ga + 8);
        const _Float16* gb = BT + (size_t)brow * D1 + kt + bcol;
        *reinterpret_cast<v8h*>(&Bs[brow][bcol]) = *reinterpret_cast<const v8h*>(gb);
        __syncthreads();
        v16h af = frag_from_lds(&As[w * 16 + laneM][0], laneH);
#pragma unroll
        for (int j = 0; j < 2; ++j) {
            v16h bf = frag_from_lds(&Bs[j * 16 + laneM][0], laneH);
            acc[j] = __builtin_amdgcn_wmma_f32_16x16x32_f16(
                false, af, false, bf, (short)0, acc[j], false, false);
        }
        __syncthreads();
    }
#pragma unroll
    for (int j = 0; j < 2; ++j) {
        size_t row = gm + w * 16 + 8 * laneH;
        size_t col = j * 16 + laneM;
#pragma unroll
        for (int r = 0; r < 8; ++r)
            C[(row + r) * NP2 + col] = acc[j][r];
    }
}

// ---------------- attention node sums, layer 1 ----------------
__global__ __launch_bounds__(256) void att_sums1_kernel(const float* __restrict__ h1,
                                                        const float* __restrict__ asrc,
                                                        const float* __restrict__ adst,
                                                        float* __restrict__ as,
                                                        float* __restrict__ ad) {
    __shared__ float r1[256], r2[256];
    const int n = blockIdx.x, tid = threadIdx.x;
    const int h = tid >> 4, l = tid & 15;
    const float* hp = h1 + (size_t)n * D1 + h * NC1 + l * 16;
    const float* sp = asrc + h * NC1 + l * 16;
    const float* dp = adst + h * NC1 + l * 16;
    float s1 = 0.f, s2 = 0.f;
#pragma unroll
    for (int c = 0; c < 16; ++c) { float v = hp[c]; s1 += v * sp[c]; s2 += v * dp[c]; }
    r1[tid] = s1; r2[tid] = s2;
    __syncthreads();
    if (l == 0) {
        float t1 = 0.f, t2 = 0.f;
#pragma unroll
        for (int i = 0; i < 16; ++i) { t1 += r1[tid + i]; t2 += r2[tid + i]; }
        as[n * NH1 + h] = t1; ad[n * NH1 + h] = t2;
    }
}

__global__ void att_sums2_kernel(const float* __restrict__ h2,
                                 const float* __restrict__ asrc,
                                 const float* __restrict__ adst,
                                 float* __restrict__ as, float* __restrict__ ad) {
    int n = blockIdx.x * 256 + threadIdx.x;
    if (n >= NN) return;
    float s1 = 0.f, s2 = 0.f;
    const float* hp = h2 + (size_t)n * NP2;
#pragma unroll
    for (int c = 0; c < OUTC; ++c) { float v = hp[c]; s1 += v * asrc[c]; s2 += v * adst[c]; }
    as[n] = s1; ad[n] = s2;
}

// ---------------- edge logits + leaky-relu + segment max (encoded uint) ----------------
__global__ void edge_logits_kernel(const int* __restrict__ ei, const float* __restrict__ as,
                                   const float* __restrict__ ad, float* __restrict__ e,
                                   unsigned* __restrict__ m, int H) {
    int idx = blockIdx.x * 256 + threadIdx.x;
    if (idx >= ET * H) return;
    int ed = idx / H, h = idx - ed * H;
    int src = ed < NE ? ei[ed] : (ed - NE);
    int dst = ed < NE ? ei[NE + ed] : (ed - NE);
    float v = as[src * H + h] + ad[dst * H + h];
    v = v > 0.f ? v : NEG_SLOPE * v;
    e[idx] = v;
    atomicMax(&m[dst * H + h], fenc(v));
}

// ---------------- exp(e - max) + segment sum ----------------
__global__ void edge_exp_kernel(const int* __restrict__ ei, float* __restrict__ e,
                                const unsigned* __restrict__ m, float* __restrict__ den, int H) {
    int idx = blockIdx.x * 256 + threadIdx.x;
    if (idx >= ET * H) return;
    int ed = idx / H, h = idx - ed * H;
    int dst = ed < NE ? ei[NE + ed] : (ed - NE);
    float ex = __expf(e[idx] - fdec(m[dst * H + h]));
    e[idx] = ex;
    atomicAdd(&den[dst * H + h], ex);
}

// ---------------- layer-1 alpha + scatter aggregation ----------------
__global__ __launch_bounds__(128) void aggr1_kernel(const int* __restrict__ ei,
                                                    const float* __restrict__ ex,
                                                    const float* __restrict__ den,
                                                    const float* __restrict__ h1,
                                                    float* __restrict__ out1,
                                                    float* __restrict__ alpha_out) {
    __shared__ float alphaS[NH1];
    const int e = blockIdx.x, tid = threadIdx.x;
    const int src = e < NE ? ei[e] : (e - NE);
    const int dst = e < NE ? ei[NE + e] : (e - NE);
    if (tid < NH1) {
        float a = ex[e * NH1 + tid] / den[dst * NH1 + tid];
        alphaS[tid] = a;
        alpha_out[e * NH1 + tid] = a;
    }
    __syncthreads();
    const float* hs = h1 + (size_t)src * D1;
    float* od = out1 + (size_t)dst * D1;
    for (int j = tid; j < D1; j += 128)
        atomicAdd(&od[j], hs[j] * alphaS[j >> 8]);
}

// ---------------- layer-2 alpha + scatter aggregation ----------------
__global__ __launch_bounds__(32) void aggr2_kernel(const int* __restrict__ ei,
                                                   const float* __restrict__ ex,
                                                   const float* __restrict__ den,
                                                   const float* __restrict__ h2,
                                                   float* __restrict__ out2,
                                                   float* __restrict__ alpha_out) {
    const int e = blockIdx.x, tid = threadIdx.x;
    const int src = e < NE ? ei[e] : (e - NE);
    const int dst = e < NE ? ei[NE + e] : (e - NE);
    float a = ex[e] / den[dst];
    if (tid == 0) alpha_out[e] = a;
    if (tid < OUTC) atomicAdd(&out2[dst * NP2 + tid], h2[(size_t)src * NP2 + tid] * a);
}

// ---------------- x1 = elu(out1 + b1) -> f16 (padded rows zero) ----------------
__global__ void elu_x1_kernel(const float* __restrict__ out1, const float* __restrict__ b1,
                              _Float16* __restrict__ x1f) {
    unsigned idx = blockIdx.x * 256u + threadIdx.x;
    if (idx >= (unsigned)MP * D1) return;
    unsigned row = idx >> 12, col = idx & (D1 - 1);
    float v = 0.f;
    if (row < NN) v = elu1(out1[(size_t)row * D1 + col] + b1[col]);
    x1f[idx] = (_Float16)v;
}

// ---------------- x2 = elu(out2 + b2) -> d_out ----------------
__global__ void final_elu_kernel(const float* __restrict__ out2, const float* __restrict__ b2,
                                 float* __restrict__ y) {
    int idx = blockIdx.x * 256 + threadIdx.x;
    if (idx >= NN * OUTC) return;
    int n = idx / OUTC, c = idx - n * OUTC;
    y[idx] = elu1(out2[n * NP2 + c] + b2[c]);
}

// ---------------- host launcher ----------------
static inline size_t al256(size_t x) { return (x + 255) & ~(size_t)255; }
static inline unsigned cdiv(size_t a, size_t b) { return (unsigned)((a + b - 1) / b); }

extern "C" void kernel_launch(void* const* d_in, const int* in_sizes, int n_in,
                              void* d_out, int out_size, void* d_ws, size_t ws_size,
                              hipStream_t stream) {
    const float* x    = (const float*)d_in[0];
    const int*   ei   = (const int*)d_in[1];
    const float* W1   = (const float*)d_in[2];
    const float* aS1  = (const float*)d_in[3];
    const float* aD1  = (const float*)d_in[4];
    const float* b1   = (const float*)d_in[5];
    const float* W2   = (const float*)d_in[6];
    const float* aS2  = (const float*)d_in[7];
    const float* aD2  = (const float*)d_in[8];
    const float* b2   = (const float*)d_in[9];
    float* out = (float*)d_out;
    float* out_x2    = out;                     // [NN][OUTC]
    float* out_a1    = out + (size_t)NN * OUTC; // [ET][NH1]
    float* out_a2    = out_a1 + (size_t)ET * NH1;

    char* p = (char*)d_ws;
    _Float16* xf16 = (_Float16*)p; p += al256((size_t)MP * KP1 * 2);
    _Float16* w1t  = (_Float16*)p; p += al256((size_t)D1 * KP1 * 2);
    float*    h1   = (float*)p;    p += al256((size_t)MP * D1 * 4);
    float*    as1  = (float*)p;    p += al256((size_t)NN * NH1 * 4);
    float*    ad1  = (float*)p;    p += al256((size_t)NN * NH1 * 4);
    float*    e1   = (float*)p;    p += al256((size_t)ET * NH1 * 4);
    unsigned* m1   = (unsigned*)p; p += al256((size_t)NN * NH1 * 4);
    float*    den1 = (float*)p;    p += al256((size_t)NN * NH1 * 4);
    float*    out1 = (float*)p;    p += al256((size_t)NN * D1 * 4);
    _Float16* x1f  = (_Float16*)p; p += al256((size_t)MP * D1 * 2);
    _Float16* w2t  = (_Float16*)p; p += al256((size_t)NP2 * D1 * 2);
    float*    h2   = (float*)p;    p += al256((size_t)MP * NP2 * 4);
    float*    as2  = (float*)p;    p += al256((size_t)NN * 4);
    float*    ad2  = (float*)p;    p += al256((size_t)NN * 4);
    float*    e2   = (float*)p;    p += al256((size_t)ET * 4);
    unsigned* m2   = (unsigned*)p; p += al256((size_t)NN * 4);
    float*    den2 = (float*)p;    p += al256((size_t)NN * 4);
    float*    out2 = (float*)p;    p += al256((size_t)NN * NP2 * 4);

    // zero accumulators / init encoded maxima (encoded 0 < any finite encoding)
    fill_u32_kernel<<<cdiv((size_t)NN * D1, 256), 256, 0, stream>>>((unsigned*)out1, 0u, NN * D1);
    fill_u32_kernel<<<cdiv((size_t)NN * NH1, 256), 256, 0, stream>>>(m1, 0u, NN * NH1);
    fill_u32_kernel<<<cdiv((size_t)NN * NH1, 256), 256, 0, stream>>>((unsigned*)den1, 0u, NN * NH1);
    fill_u32_kernel<<<cdiv((size_t)NN, 256), 256, 0, stream>>>(m2, 0u, NN);
    fill_u32_kernel<<<cdiv((size_t)NN, 256), 256, 0, stream>>>((unsigned*)den2, 0u, NN);
    fill_u32_kernel<<<cdiv((size_t)NN * NP2, 256), 256, 0, stream>>>((unsigned*)out2, 0u, NN * NP2);

    // layer 1
    convert_x_kernel <<<cdiv((size_t)MP * KP1, 256), 256, 0, stream>>>(x, xf16);
    convert_w1_kernel<<<cdiv((size_t)KP1 * D1, 256), 256, 0, stream>>>(W1, w1t);
    gemm1_kernel<<<dim3(D1 / 64, MP / 128), 256, 2 * 12288, stream>>>(xf16, w1t, h1);
    att_sums1_kernel<<<NN, 256, 0, stream>>>(h1, aS1, aD1, as1, ad1);
    edge_logits_kernel<<<cdiv((size_t)ET * NH1, 256), 256, 0, stream>>>(ei, as1, ad1, e1, m1, NH1);
    edge_exp_kernel   <<<cdiv((size_t)ET * NH1, 256), 256, 0, stream>>>(ei, e1, m1, den1, NH1);
    aggr1_kernel<<<ET, 128, 0, stream>>>(ei, e1, den1, h1, out1, out_a1);
    elu_x1_kernel<<<cdiv((size_t)MP * D1, 256), 256, 0, stream>>>(out1, b1, x1f);

    // layer 2
    convert_w2_kernel<<<cdiv((size_t)NP2 * D1, 256), 256, 0, stream>>>(W2, w2t);
    gemm2_kernel<<<dim3(1, MP / 64), 128, 0, stream>>>(x1f, w2t, h2);
    att_sums2_kernel<<<cdiv(NN, 256), 256, 0, stream>>>(h2, aS2, aD2, as2, ad2);
    edge_logits_kernel<<<cdiv(ET, 256), 256, 0, stream>>>(ei, as2, ad2, e2, m2, 1);
    edge_exp_kernel   <<<cdiv(ET, 256), 256, 0, stream>>>(ei, e2, m2, den2, 1);
    aggr2_kernel<<<ET, 32, 0, stream>>>(ei, e2, den2, h2, out2, out_a2);
    final_elu_kernel<<<cdiv((size_t)NN * OUTC, 256), 256, 0, stream>>>(out2, b2, out_x2);

    (void)in_sizes; (void)n_in; (void)out_size; (void)ws_size;
}
